// DeformConv2dSuccess_83734682402951
// MI455X (gfx1250) — matrix operand order repaired and checked by reading, not compiled, not run on hardware
//
#include <hip/hip_runtime.h>
#include <hip/hip_bf16.h>
#include <stdint.h>

// ---------------------------------------------------------------------------
// Deformable Conv2d (DCNv2) for MI455X / gfx1250.
// B=8, C=256, O=256, H=W=64, K taps = 9.  Main GEMM: M=256, K=2304, N=32768
// via v_wmma_f32_16x16x32_f16, f32 accumulate, async-to-LDS double buffering.
// ---------------------------------------------------------------------------

typedef __attribute__((ext_vector_type(16))) _Float16 v16h;
typedef __attribute__((ext_vector_type(8)))  float    v8f;
typedef __attribute__((ext_vector_type(4)))  int      vec4i;

#define CIN   256
#define OCH   256
#define HH    64
#define WW    64
#define HW    4096         // 64*64
#define BATCH 8
#define KTAP  9
#define KDIM  2304         // CIN * KTAP
#define NDIM  32768        // BATCH * HW

// ---------------------------------------------------------------------------
// Kernel 1: fused 27-channel 3x3 conv (18 offset channels + 9 mask channels,
// sigmoid applied to mask).  One thread per pixel computes all 27 channels so
// each x value is loaded exactly once per tap; weights are block-uniform ->
// scalar loads.  x (32MB) is L2-resident.
// om layout: [b][ch 0..26][hw], f32.
// ---------------------------------------------------------------------------
__global__ __launch_bounds__(256) void conv_offs_mask(
    const float* __restrict__ x, const float* __restrict__ wp,
    const float* __restrict__ wm, float* __restrict__ om) {
  int tid = blockIdx.x * 256 + threadIdx.x;        // 0..32767
  int hw  = tid & (HW - 1);
  int b   = tid >> 12;
  int i   = hw >> 6;
  int j   = hw & 63;

  float acc[27];
#pragma unroll
  for (int ch = 0; ch < 27; ++ch) acc[ch] = 0.f;

  const float* xb = x + (size_t)b * CIN * HW;
  for (int c = 0; c < CIN; ++c) {
    const float* xc = xb + (size_t)c * HW;
#pragma unroll
    for (int u = 0; u < 3; ++u) {
      int y = i + u - 1;
      bool yv = (unsigned)y < (unsigned)HH;
#pragma unroll
      for (int v = 0; v < 3; ++v) {
        int xx = j + v - 1;
        bool xv_ok = yv && ((unsigned)xx < (unsigned)WW);
        float xv = xv_ok ? xc[y * WW + xx] : 0.f;
        int wi = c * 9 + u * 3 + v;
#pragma unroll
        for (int ch = 0; ch < 18; ++ch) acc[ch]      += xv * wp[(size_t)ch * KDIM + wi];
#pragma unroll
        for (int ch = 0; ch < 9;  ++ch) acc[18 + ch] += xv * wm[(size_t)ch * KDIM + wi];
      }
    }
  }

  float* ob = om + (size_t)b * 27 * HW + hw;
#pragma unroll
  for (int ch = 0; ch < 18; ++ch) ob[(size_t)ch * HW] = acc[ch];
#pragma unroll
  for (int ch = 18; ch < 27; ++ch)
    ob[(size_t)ch * HW] = 1.f / (1.f + __expf(-acc[ch]));
}

// ---------------------------------------------------------------------------
// Kernel 2: deformable im2col.  One block per (b, kk, pixel): offsets/mask are
// block-uniform (scalar loads + scalar branches); 256 lanes = input channels.
// Writes colT[n][kk*256 + c] as f16 -> 512B contiguous stores per block.
// ---------------------------------------------------------------------------
__global__ __launch_bounds__(256) void deform_im2col(
    const float* __restrict__ x, const float* __restrict__ om,
    _Float16* __restrict__ colT) {
  int bid = blockIdx.x;                     // b*9*4096 + kk*4096 + hw
  int hw  = bid & (HW - 1);
  int kk  = (bid >> 12) % KTAP;
  int b   = bid / (KTAP * HW);
  int i   = hw >> 6;
  int j   = hw & 63;

  const float* omb = om + (size_t)b * 27 * HW;
  float dy = omb[(size_t)(2 * kk)     * HW + hw];
  float dx = omb[(size_t)(2 * kk + 1) * HW + hw];
  float mk = omb[(size_t)(18 + kk)    * HW + hw];

  float py = (float)(i - 1 + kk / 3) + dy;
  float px = (float)(j - 1 + kk % 3) + dx;
  float y0f = floorf(py), x0f = floorf(px);
  int   y0  = (int)y0f,   x0  = (int)x0f;
  float wy1 = py - y0f,   wx1 = px - x0f;
  float wy0 = 1.f - wy1,  wx0 = 1.f - wx1;

  bool y0v = (unsigned)y0       < (unsigned)HH;
  bool y1v = (unsigned)(y0 + 1) < (unsigned)HH;
  bool x0v = (unsigned)x0       < (unsigned)WW;
  bool x1v = (unsigned)(x0 + 1) < (unsigned)WW;

  int c = threadIdx.x;
  const float* xc = x + ((size_t)b * CIN + c) * HW;
  float v = 0.f;
  if (y0v && x0v) v += wy0 * wx0 * xc[y0 * WW + x0];
  if (y0v && x1v) v += wy0 * wx1 * xc[y0 * WW + x0 + 1];
  if (y1v && x0v) v += wy1 * wx0 * xc[(y0 + 1) * WW + x0];
  if (y1v && x1v) v += wy1 * wx1 * xc[(y0 + 1) * WW + x0 + 1];

  size_t n = (size_t)b * HW + hw;
  colT[n * KDIM + kk * CIN + c] = (_Float16)(v * mk);
}

// ---------------------------------------------------------------------------
// Kernel 3: repack w_d (O,C,3,3) f32 -> Wh[o][kk*256 + c] f16 (row-major K).
// ---------------------------------------------------------------------------
__global__ __launch_bounds__(256) void wconv(
    const float* __restrict__ wd, _Float16* __restrict__ Wh) {
  int t = blockIdx.x * 256 + threadIdx.x;
  if (t >= OCH * KDIM) return;
  int o  = t / KDIM;
  int r  = t - o * KDIM;       // kk*256 + c
  int kk = r >> 8;
  int c  = r & 255;
  Wh[t] = (_Float16)wd[((size_t)o * CIN + c) * KTAP + kk];
}

// ---------------------------------------------------------------------------
// Kernel 4: WMMA GEMM  out[o][n] = sum_k Wh[o][k] * colT[n][k] + bias[o].
// 128x128 tile per WG, 8 waves (2 in M x 4 in N), 64x32 per wave = 4x2
// 16x16 accumulator tiles.  K stepped by 32.  Double-buffered LDS filled with
// GLOBAL_LOAD_ASYNC_TO_LDS_B128 (ASYNCcnt) when the toolchain exposes it;
// otherwise falls back to register-staged sync copies.
// ---------------------------------------------------------------------------
#define MT 128
#define NT 128
#define KB 32
#define SA 40                      // padded LDS row stride in halves
#define BUFH ((MT + NT) * SA)      // halves per double-buffer slot
#define ITERS (KDIM / KB)          // 72

#if __has_builtin(__builtin_amdgcn_global_load_async_to_lds_b128) && \
    __has_builtin(__builtin_amdgcn_s_wait_asynccnt)
#define HAVE_ASYNC 1
#else
#define HAVE_ASYNC 0
#endif

__device__ __forceinline__ v16h frag16(const _Float16* p0, const _Float16* p1) {
  struct Q { uint4 a, b; } q;
  q.a = *(const uint4*)p0;
  q.b = *(const uint4*)p1;
  return __builtin_bit_cast(v16h, q);
}

#if HAVE_ASYNC
typedef __attribute__((address_space(1))) vec4i* gptr_v4i;   // "__device__" AS
typedef __attribute__((address_space(3))) vec4i* lptr_v4i;   // "__shared__" AS

__device__ __forceinline__ void async_ld16(const _Float16* g, _Float16* l) {
  __builtin_amdgcn_global_load_async_to_lds_b128(
      (gptr_v4i)(void*)g, (lptr_v4i)(void*)l, 0, 0);
}
#endif

__global__ __launch_bounds__(256) void wmma_gemm(
    const _Float16* __restrict__ Wh, const _Float16* __restrict__ colT,
    const float* __restrict__ bias, float* __restrict__ out) {
#if HAVE_ASYNC
  __shared__ __align__(16) _Float16 lds[2 * BUFH];
#else
  __shared__ __align__(16) _Float16 lds[BUFH];
#endif

  int tile  = blockIdx.x;                 // 512 tiles
  int mBase = (tile & 1) * MT;            // 2 M-tiles
  int nBase = (tile >> 1) * NT;           // 256 N-tiles
  int t     = threadIdx.x;
  int lane  = t & 31;
  int wave  = t >> 5;
  int wm    = (wave & 1) * 64;            // wave M offset in tile
  int wn    = (wave >> 1) * 32;           // wave N offset in tile
  int lg    = lane >> 4;                  // lane group 0/1
  int lr    = lane & 15;

  // Per-thread staging coordinates: 4 uint4 rows-of-8-halves per buffer.
  int e0 = t, e1 = t + 256;
  int rowA0 = e0 >> 2, chA0 = e0 & 3;
  int rowA1 = e1 >> 2, chA1 = e1 & 3;

  v8f acc[4][2];
#pragma unroll
  for (int a = 0; a < 4; ++a)
#pragma unroll
    for (int bb = 0; bb < 2; ++bb) acc[a][bb] = (v8f){};

#if HAVE_ASYNC
  // ---- async double-buffered pipeline ----
  // issue(buf, kt): 4 async b128 per thread (2 for A, 2 for B).
  auto issue = [&](int buf, int kt) {
    _Float16* dA = lds + buf * BUFH;
    _Float16* dB = dA + MT * SA;
    async_ld16(Wh   + (size_t)(mBase + rowA0) * KDIM + kt + chA0 * 8,
               dA + rowA0 * SA + chA0 * 8);
    async_ld16(Wh   + (size_t)(mBase + rowA1) * KDIM + kt + chA1 * 8,
               dA + rowA1 * SA + chA1 * 8);
    async_ld16(colT + (size_t)(nBase + rowA0) * KDIM + kt + chA0 * 8,
               dB + rowA0 * SA + chA0 * 8);
    async_ld16(colT + (size_t)(nBase + rowA1) * KDIM + kt + chA1 * 8,
               dB + rowA1 * SA + chA1 * 8);
  };

  issue(0, 0);
  issue(1, KB);
  __builtin_amdgcn_s_wait_asynccnt(4);    // buf0 (own ops) landed
  __syncthreads();                        // buf0 landed for all waves

  int cur = 0;
  for (int it = 0; it < ITERS; ++it) {
    const _Float16* bA = lds + cur * BUFH;
    const _Float16* bB = bA + MT * SA;

    // Pull this k-step's fragments into registers.
    v16h aF[4], bF[2];
#pragma unroll
    for (int s = 0; s < 4; ++s) {
      const _Float16* base = bA + (wm + s * 16 + lr) * SA;
      aF[s] = frag16(base + lg * 8, base + 16 + lg * 8);
    }
#pragma unroll
    for (int s = 0; s < 2; ++s) {
      const _Float16* base = bB + (wn + s * 16 + lr) * SA;
      bF[s] = frag16(base + lg * 8, base + 16 + lg * 8);
    }
    __syncthreads();                      // all waves done reading buf[cur]

    // Refill buf[cur] for step it+2 while the WMMAs run.
    if (it + 2 < ITERS) issue(cur, (it + 2) * KB);

#pragma unroll
    for (int a = 0; a < 4; ++a)
#pragma unroll
      for (int bb = 0; bb < 2; ++bb)
        acc[a][bb] = __builtin_amdgcn_wmma_f32_16x16x32_f16(
            false, aF[a], false, bF[bb], (short)0, acc[a][bb], false, false);

    // Wait only for the iteration-old ops (buf[nxt]); fresh ones stay in flight.
    if (it + 2 < ITERS) __builtin_amdgcn_s_wait_asynccnt(4);
    else                __builtin_amdgcn_s_wait_asynccnt(0);
    __syncthreads();                      // buf[nxt] landed for all waves
    cur ^= 1;
  }
#else
  // ---- fallback: register-staged sync copies, single buffer ----
  _Float16* ldsA = lds;
  _Float16* ldsB = lds + MT * SA;
  for (int kt = 0; kt < KDIM; kt += KB) {
    __syncthreads();
    *(uint4*)(ldsA + rowA0 * SA + chA0 * 8) =
        *(const uint4*)(Wh + (size_t)(mBase + rowA0) * KDIM + kt + chA0 * 8);
    *(uint4*)(ldsA + rowA1 * SA + chA1 * 8) =
        *(const uint4*)(Wh + (size_t)(mBase + rowA1) * KDIM + kt + chA1 * 8);
    *(uint4*)(ldsB + rowA0 * SA + chA0 * 8) =
        *(const uint4*)(colT + (size_t)(nBase + rowA0) * KDIM + kt + chA0 * 8);
    *(uint4*)(ldsB + rowA1 * SA + chA1 * 8) =
        *(const uint4*)(colT + (size_t)(nBase + rowA1) * KDIM + kt + chA1 * 8);
    __syncthreads();

    v16h aF[4], bF[2];
#pragma unroll
    for (int s = 0; s < 4; ++s) {
      const _Float16* base = ldsA + (wm + s * 16 + lr) * SA;
      aF[s] = frag16(base + lg * 8, base + 16 + lg * 8);
    }
#pragma unroll
    for (int s = 0; s < 2; ++s) {
      const _Float16* base = ldsB + (wn + s * 16 + lr) * SA;
      bF[s] = frag16(base + lg * 8, base + 16 + lg * 8);
    }
#pragma unroll
    for (int a = 0; a < 4; ++a)
#pragma unroll
      for (int bb = 0; bb < 2; ++bb)
        acc[a][bb] = __builtin_amdgcn_wmma_f32_16x16x32_f16(
            false, aF[a], false, bF[bb], (short)0, acc[a][bb], false, false);
  }
#endif

  // Epilogue: C/D layout -> VGPR v: M = base + lg*8 + v, N = lr.  Add bias,
  // scatter to NCHW: out[((n>>12)*256 + m)*4096 + (n&4095)].
#pragma unroll
  for (int a = 0; a < 4; ++a)
#pragma unroll
    for (int bb = 0; bb < 2; ++bb) {
      int m0 = mBase + wm + a * 16 + lg * 8;
      int n  = nBase + wn + bb * 16 + lr;
      int bo = n >> 12;
      int hw = n & (HW - 1);
      float* op = out + (size_t)bo * OCH * HW + hw;
#pragma unroll
      for (int v = 0; v < 8; ++v) {
        int m = m0 + v;
        op[(size_t)m * HW] = acc[a][bb][v] + bias[m];
      }
    }
}

// ---------------------------------------------------------------------------
// Launch
// ---------------------------------------------------------------------------
extern "C" void kernel_launch(void* const* d_in, const int* in_sizes, int n_in,
                              void* d_out, int out_size, void* d_ws, size_t ws_size,
                              hipStream_t stream) {
  (void)in_sizes; (void)n_in; (void)out_size; (void)ws_size;
  const float* x   = (const float*)d_in[0];
  const float* w_p = (const float*)d_in[1];
  const float* w_m = (const float*)d_in[2];
  const float* w_d = (const float*)d_in[3];
  const float* b_d = (const float*)d_in[4];
  float* out = (float*)d_out;

  char* ws = (char*)d_ws;
  float*    om   = (float*)(ws);                               // 8*27*4096 f32  (~3.5 MB)
  _Float16* Wh   = (_Float16*)(ws + (size_t)15 * 1024 * 1024); // 256*2304 f16   (~1.2 MB)
  _Float16* colT = (_Float16*)(ws + (size_t)17 * 1024 * 1024); // 32768*2304 f16 (~151 MB)

  // 1) offsets + mask conv (32768 pixels, 256 threads/block)
  conv_offs_mask<<<BATCH * HW / 256, 256, 0, stream>>>(x, w_p, w_m, om);
  // 2) deformable im2col (one block per (b,kk,pixel))
  deform_im2col<<<BATCH * KTAP * HW, 256, 0, stream>>>(x, om, colT);
  // 3) weight repack to f16
  wconv<<<(OCH * KDIM + 255) / 256, 256, 0, stream>>>(w_d, Wh);
  // 4) WMMA GEMM: 2 M-tiles x 256 N-tiles = 512 workgroups
  wmma_gemm<<<(OCH / MT) * (NDIM / NT), 256, 0, stream>>>(Wh, colT, b_d, out);
}
